// Attention2d_54820962566679
// MI455X (gfx1250) — matrix operand order repaired and checked by reading, not compiled, not run on hardware
//
#include <hip/hip_runtime.h>

typedef __attribute__((ext_vector_type(16))) __bf16 v16bf;
typedef __attribute__((ext_vector_type(8)))  float  v8f;
typedef __attribute__((ext_vector_type(4)))  unsigned int v4u;
typedef __attribute__((ext_vector_type(8)))  int v8i_;
typedef __attribute__((ext_vector_type(4)))  int v4i_;

union Frag { v16bf v; unsigned u[8]; };

__device__ __forceinline__ unsigned short f2bf(float f) {
    union { __bf16 h; unsigned short u; } c;
    c.h = (__bf16)f;                 // native RNE convert on gfx1250
    return c.u;
}

// Problem constants
constexpr int BATCH = 256;
constexpr int NTOK  = 196;
constexpr int XDIM  = 384;
constexpr int HEADS = 8;
constexpr int KDIM  = 32;
constexpr int VDIM  = 128;
constexpr int HID   = 1536;          // (2*32+128)*8
constexpr int NPAD  = 224;           // key/row padding to multiple of 32
constexpr int ROWS  = BATCH * NTOK;  // 50176 = 392*128

// Workspace byte offsets
constexpr size_t Q_OFF  = 0;
constexpr size_t K_OFF  = (size_t)2048 * NPAD * KDIM * 2;                 // 29,360,128
constexpr size_t V_OFF  = K_OFF + (size_t)2048 * 16 * NPAD * 2 * 2;       // 58,720,256
constexpr size_t AO_OFF = V_OFF + (size_t)2048 * (NPAD/2) * VDIM * 2 * 2; // 176,160,768
constexpr size_t B_OFF  = AO_OFF + (size_t)ROWS * 1024 * 2;               // 278,921,216
constexpr size_t V_BYTES = AO_OFF - V_OFF;

#if defined(__has_builtin)
#if __has_builtin(__builtin_amdgcn_tensor_load_to_lds)
#define HAVE_TDM 1
#endif
#endif

// ---------------------------------------------------------------------------
// Kernel 1: expand relative-position bias table -> [H,196,196] f32
// ---------------------------------------------------------------------------
__global__ __launch_bounds__(256)
void bias_kernel(const float* __restrict__ ab, const int* __restrict__ idxs,
                 float* __restrict__ bias, int n_off) {
    int i = blockIdx.x * 256 + threadIdx.x;
    if (i >= NTOK * NTOK) return;
    int id = idxs[i];
    #pragma unroll
    for (int h = 0; h < HEADS; ++h)
        bias[(size_t)h * NTOK * NTOK + i] = ab[h * n_off + id];
}

// ---------------------------------------------------------------------------
// Kernel 2: qkv = x @ Wqkv^T + bqkv  (bf16 WMMA, double-buffered LDS staging,
//   scatter into q/kT/vP layouts)
//   q  : [BH][NPAD][32]              bf16
//   kT : [BH][16][NPAD][2]           bf16 (pair-interleaved along kd)
//   vP : [BH][NPAD/2][128][2]        bf16 (pair-interleaved along token m)
// ---------------------------------------------------------------------------
__global__ __launch_bounds__(256)
void qkv_kernel(const float* __restrict__ x, const float* __restrict__ Wqkv,
                const float* __restrict__ bqkv,
                unsigned short* __restrict__ qbuf,
                unsigned short* __restrict__ kbuf,
                unsigned short* __restrict__ vbuf) {
    __shared__ unsigned short As[2][128 * 32];      // [row][k]
    __shared__ unsigned short Bs[2][16 * 128 * 2];  // [kpair][col][2]
    const int t = threadIdx.x;
    const int lane = t & 31, w = t >> 5;
    const int wm = w & 3, wn = w >> 2;              // waves: 4 (M) x 2 (N)
    const int row0 = blockIdx.x * 128;
    const int col0 = blockIdx.y * 128;
    const int lx = lane & 15, lh = lane >> 4;

    v8f acc[2][4];
    #pragma unroll
    for (int i = 0; i < 2; ++i)
        #pragma unroll
        for (int j = 0; j < 4; ++j) acc[i][j] = v8f{};

    auto stage = [&](int buf, int kt) {
        int r = t >> 1, half = t & 1;
        const float* srcA = x + (size_t)(row0 + r) * XDIM + kt + half * 16;
        unsigned short* dstA = As[buf] + r * 32 + half * 16;
        #pragma unroll
        for (int i = 0; i < 16; ++i) dstA[i] = f2bf(srcA[i]);
        const float* srcB = Wqkv + (size_t)(col0 + r) * XDIM + kt + half * 16;
        #pragma unroll
        for (int i = 0; i < 16; ++i) {
            int k = half * 16 + i;
            Bs[buf][((k >> 1) * 128 + r) * 2 + (k & 1)] = f2bf(srcB[i]);
        }
    };

    stage(0, 0);
    __syncthreads();

    for (int kt = 0; kt < XDIM; kt += 32) {
        int cur = (kt >> 5) & 1, nxt = cur ^ 1;
        if (kt + 32 < XDIM) stage(nxt, kt + 32);

        Frag a[2], b[4];
        #pragma unroll
        for (int mt = 0; mt < 2; ++mt) {
            int r = wm * 32 + mt * 16 + lx;
            #pragma unroll
            for (int j = 0; j < 8; ++j) {
                int k0 = lh * 16 + 2 * j;
                a[mt].u[j] = *(const unsigned*)(As[cur] + r * 32 + k0);
            }
        }
        #pragma unroll
        for (int nt = 0; nt < 4; ++nt) {
            int n = wn * 64 + nt * 16 + lx;
            #pragma unroll
            for (int j = 0; j < 8; ++j) {
                int kp = j + lh * 8;
                b[nt].u[j] = *(const unsigned*)(Bs[cur] + (kp * 128 + n) * 2);
            }
        }
        #pragma unroll
        for (int mt = 0; mt < 2; ++mt)
            #pragma unroll
            for (int nt = 0; nt < 4; ++nt)
                acc[mt][nt] = __builtin_amdgcn_wmma_f32_16x16x32_bf16(
                    false, a[mt].v, false, b[nt].v, (short)0, acc[mt][nt], false, false);
        __syncthreads();
    }

    // epilogue: hoist per-column and per-row index math, then scatter bf16
    int   hh[4], rrv[4];
    float bq[4];
    #pragma unroll
    for (int nt = 0; nt < 4; ++nt) {
        int c = col0 + wn * 64 + nt * 16 + lx;
        bq[nt]  = bqkv[c];
        hh[nt]  = c / 192;
        rrv[nt] = c - hh[nt] * 192;
    }
    #pragma unroll
    for (int mt = 0; mt < 2; ++mt) {
        #pragma unroll
        for (int j = 0; j < 8; ++j) {
            int gr = row0 + wm * 32 + mt * 16 + j + lh * 8;   // < 50176
            int bi = gr / NTOK, n = gr - bi * NTOK;
            #pragma unroll
            for (int nt = 0; nt < 4; ++nt) {
                size_t bh = (size_t)bi * HEADS + hh[nt];
                int rr = rrv[nt];
                unsigned short bf = f2bf(acc[mt][nt][j] + bq[nt]);
                if (rr < KDIM) {
                    qbuf[bh * (NPAD * KDIM) + (size_t)n * KDIM + rr] = bf;
                } else if (rr < 2 * KDIM) {
                    int kd = rr - KDIM;
                    kbuf[bh * (16 * NPAD * 2) + (size_t)((kd >> 1) * NPAD + n) * 2 + (kd & 1)] = bf;
                } else {
                    int d = rr - 2 * KDIM;
                    vbuf[bh * ((NPAD / 2) * VDIM * 2) + (size_t)((n >> 1) * VDIM + d) * 2 + (n & 1)] = bf;
                }
            }
        }
    }
}

// ---------------------------------------------------------------------------
// Kernel 3: fused attention, one wave per (b,h,16-row tile), flash softmax.
// ---------------------------------------------------------------------------
__global__ __launch_bounds__(256)
void attn_kernel(const unsigned short* __restrict__ qbuf,
                 const unsigned short* __restrict__ kbuf,
                 const unsigned short* __restrict__ vbuf,
                 const float* __restrict__ bias,
                 unsigned short* __restrict__ ao) {
    __shared__ unsigned short Ps[8 * 16 * 32];   // per-wave 16x32 bf16 P tile
    const int t = threadIdx.x;
    const int lane = t & 31, w = t >> 5;
    const int gw = blockIdx.x * 8 + w;           // 0 .. 26623
    const int bh = gw / 13;
    const int mt = gw - bh * 13;
    const int b = bh >> 3, h = bh & 7;
    const int n0 = mt * 16;
    const int lx = lane & 15, lh = lane >> 4;
    const float scale = 0.17677669529663687f;    // 1/sqrt(32)

    const unsigned short* q  = qbuf + (size_t)bh * (NPAD * KDIM);
    const unsigned short* kT = kbuf + (size_t)bh * (16 * NPAD * 2);
    const unsigned short* vP = vbuf + (size_t)bh * ((NPAD / 2) * VDIM * 2);
    const float* hb = bias + (size_t)h * NTOK * NTOK;
    unsigned short* pl = Ps + w * 16 * 32;

    Frag qa;
    {
        int r = n0 + lx;
        #pragma unroll
        for (int j = 0; j < 8; ++j) {
            int k0 = lh * 16 + 2 * j;
            qa.u[j] = *(const unsigned*)(q + r * KDIM + k0);
        }
    }

    v8f o[8];
    #pragma unroll
    for (int dt = 0; dt < 8; ++dt) o[dt] = v8f{};
    float rmax[8], rsum[8];
    #pragma unroll
    for (int j = 0; j < 8; ++j) { rmax[j] = -1e30f; rsum[j] = 0.0f; }

    for (int jt = 0; jt < 7; ++jt) {
        if (jt + 1 < 7) {   // prefetch next supertile of K^T and V
            __builtin_prefetch(kT + ((lh * 8) * NPAD + (jt + 1) * 32 + lx) * 2, 0, 1);
            __builtin_prefetch(vP + (((jt + 1) * 16 + lh * 8) * VDIM + lx) * 2, 0, 1);
        }
        v8f s[2];
        #pragma unroll
        for (int sub = 0; sub < 2; ++sub) {
            int c0 = jt * 32 + sub * 16;
            Frag kb;
            #pragma unroll
            for (int j = 0; j < 8; ++j) {
                int kp = j + lh * 8;
                kb.u[j] = *(const unsigned*)(kT + (kp * NPAD + c0 + lx) * 2);
            }
            v8f zero = v8f{};
            s[sub] = __builtin_amdgcn_wmma_f32_16x16x32_bf16(
                false, qa.v, false, kb.v, (short)0, zero, false, false);
        }
        int col0 = jt * 32 + lx;
        int col1 = col0 + 16;
        #pragma unroll
        for (int j = 0; j < 8; ++j) {
            int r2 = j + lh * 8;
            int rg = n0 + r2;
            int br = rg < NTOK ? rg : NTOK - 1;
            float v0 = s[0][j] * scale + hb[(size_t)br * NTOK + (col0 < NTOK ? col0 : 0)];
            float v1 = s[1][j] * scale + hb[(size_t)br * NTOK + (col1 < NTOK ? col1 : 0)];
            if (col0 >= NTOK) v0 = -1e30f;
            if (col1 >= NTOK) v1 = -1e30f;
            float tmax = fmaxf(v0, v1);
            #pragma unroll
            for (int m = 1; m < 16; m <<= 1) tmax = fmaxf(tmax, __shfl_xor(tmax, m, 32));
            float nm = fmaxf(rmax[j], tmax);
            float corr = __expf(rmax[j] - nm);
            float p0 = __expf(v0 - nm);
            float p1 = __expf(v1 - nm);
            float tsum = p0 + p1;
            #pragma unroll
            for (int m = 1; m < 16; m <<= 1) tsum += __shfl_xor(tsum, m, 32);
            rsum[j] = rsum[j] * corr + tsum;
            rmax[j] = nm;
            #pragma unroll
            for (int dt = 0; dt < 8; ++dt) o[dt][j] *= corr;
            pl[r2 * 32 + lx] = f2bf(p0);
            pl[r2 * 32 + 16 + lx] = f2bf(p1);
        }
        // wave-local: make P stores visible to the transposed frag reload
        asm volatile("s_wait_dscnt 0" ::: "memory");
        Frag pa;
        #pragma unroll
        for (int j = 0; j < 8; ++j) {
            int k0 = lh * 16 + 2 * j;
            pa.u[j] = *(const unsigned*)(pl + lx * 32 + k0);
        }
        #pragma unroll
        for (int dt = 0; dt < 8; ++dt) {
            Frag vb;
            #pragma unroll
            for (int j = 0; j < 8; ++j) {
                int mp = jt * 16 + j + lh * 8;
                vb.u[j] = *(const unsigned*)(vP + (mp * VDIM + dt * 16 + lx) * 2);
            }
            o[dt] = __builtin_amdgcn_wmma_f32_16x16x32_bf16(
                false, pa.v, false, vb.v, (short)0, o[dt], false, false);
        }
    }

    #pragma unroll
    for (int j = 0; j < 8; ++j) {
        float inv = 1.0f / rsum[j];
        int n = n0 + j + lh * 8;
        if (n < NTOK) {
            size_t ro = ((size_t)b * NTOK + n) * 1024 + h * VDIM;
            #pragma unroll
            for (int dt = 0; dt < 8; ++dt)
                ao[ro + dt * 16 + lx] = f2bf(o[dt][j] * inv);
        }
    }
}

// ---------------------------------------------------------------------------
// Kernel 4: out = attn_out @ Wproj^T + bproj (bf16 WMMA, f32 out).
// A tile is already bf16: staged alternately by the Tensor Data Mover
// (2D D#, one issue per workgroup, TENSORcnt) and per-lane async global->LDS
// DMA (ASYNCcnt). Double-buffered LDS overlaps staging with WMMA.
// ---------------------------------------------------------------------------
__global__ __launch_bounds__(256)
void proj_kernel(const unsigned short* __restrict__ a, const float* __restrict__ Wproj,
                 const float* __restrict__ bproj, float* __restrict__ out) {
    __shared__ unsigned short As[2][128 * 32];
    __shared__ unsigned short Bs[2][16 * 128 * 2];
    const int t = threadIdx.x;
    const int lane = t & 31, w = t >> 5;
    const int wm = w & 3, wn = w >> 2;
    const int row0 = blockIdx.x * 128;
    const int col0 = blockIdx.y * 128;
    const int lx = lane & 15, lh = lane >> 4;

    v8f acc[2][4];
    #pragma unroll
    for (int i = 0; i < 2; ++i)
        #pragma unroll
        for (int j = 0; j < 4; ++j) acc[i][j] = v8f{};

    // stage A tile (128 rows x 32 k bf16) into buffer `buf` for chunk kt
    auto stageA = [&](int buf, int kt) {
#ifdef HAVE_TDM
        if (buf == 0) {
            if (w == 0) {   // one TDM descriptor per workgroup
                unsigned long long ga =
                    (unsigned long long)(a + (size_t)row0 * 1024 + kt);
                unsigned lds = (unsigned)(unsigned long long)(As[0]);
                v4u g0 = { 1u,                                   // count=1
                           lds,                                  // lds_addr
                           (unsigned)ga,                         // gaddr[31:0]
                           (unsigned)((ga >> 32) & 0x01FFFFFFu) | 0x80000000u }; // type=2
                v8i_ g1 = { 0x00010000,                          // data_size=2B
                            0x04000000,                          // tensor_dim0=1024 (lo16<<16)
                            (int)0xC4000000,                     // tensor_dim1=50176 (lo16<<16)
                            0x00200000,                          // tile_dim0=32 (<<16)
                            128,                                 // tile_dim1=128
                            1024,                                // dim0 stride = 1024 elems
                            0, 0 };
                v4i_ g2 = { 1, 0, 0, 0 };                        // tensor_dim2=1 (unused)
                v4i_ g3 = { 0, 0, 0, 0 };
                v8i_ g4 = { 0, 0, 0, 0, 0, 0, 0, 0 };            // unused extra group
                __builtin_amdgcn_tensor_load_to_lds(g0, g1, g2, g3, g4, 0);
            }
            return;
        }
#endif
        #pragma unroll
        for (int i = 0; i < 2; ++i) {
            int u = t + 256 * i;
            int r = u >> 2, qq = u & 3;           // 4 x 16B per 64B row
            unsigned long long g =
                (unsigned long long)(a + (size_t)(row0 + r) * 1024 + kt + qq * 8);
            unsigned lds = (unsigned)(unsigned long long)(As[buf] + r * 32 + qq * 8);
            asm volatile("global_load_async_to_lds_b128 %0, %1, off"
                         :: "v"(lds), "v"(g) : "memory");
        }
    };
    auto stageB = [&](int buf, int kt) {
        int n = t >> 1, half = t & 1;
        const float* src = Wproj + (size_t)(col0 + n) * 1024 + kt + half * 16;
        #pragma unroll
        for (int i = 0; i < 16; ++i) {
            int k = half * 16 + i;
            Bs[buf][((k >> 1) * 128 + n) * 2 + (k & 1)] = f2bf(src[i]);
        }
    };
    auto waitA = [&](int buf) {
#ifdef HAVE_TDM
        if (buf == 0) {
#if __has_builtin(__builtin_amdgcn_s_wait_tensorcnt)
            if (w == 0) __builtin_amdgcn_s_wait_tensorcnt(0);
#else
            if (w == 0) asm volatile("s_wait_tensorcnt 0" ::: "memory");
#endif
            return;
        }
#endif
        asm volatile("s_wait_asynccnt 0" ::: "memory");
    };

    stageA(0, 0);
    stageB(0, 0);
    waitA(0);
    __syncthreads();

    for (int kt = 0; kt < 1024; kt += 32) {
        int cur = (kt >> 5) & 1, nxt = cur ^ 1;
        bool have_next = (kt + 32 < 1024);
        if (have_next) { stageA(nxt, kt + 32); stageB(nxt, kt + 32); }

        Frag fa[2], fb[4];
        #pragma unroll
        for (int mtt = 0; mtt < 2; ++mtt) {
            int r = wm * 32 + mtt * 16 + lx;
            #pragma unroll
            for (int j = 0; j < 8; ++j) {
                int k0 = lh * 16 + 2 * j;
                fa[mtt].u[j] = *(const unsigned*)(As[cur] + r * 32 + k0);
            }
        }
        #pragma unroll
        for (int nt = 0; nt < 4; ++nt) {
            int n = wn * 64 + nt * 16 + lx;
            #pragma unroll
            for (int j = 0; j < 8; ++j) {
                int kp = j + lh * 8;
                fb[nt].u[j] = *(const unsigned*)(Bs[cur] + (kp * 128 + n) * 2);
            }
        }
        #pragma unroll
        for (int mtt = 0; mtt < 2; ++mtt)
            #pragma unroll
            for (int nt = 0; nt < 4; ++nt)
                acc[mtt][nt] = __builtin_amdgcn_wmma_f32_16x16x32_bf16(
                    false, fa[mtt].v, false, fb[nt].v, (short)0, acc[mtt][nt], false, false);

        if (have_next) waitA(nxt);
        __syncthreads();
    }

    #pragma unroll
    for (int nt = 0; nt < 4; ++nt) {
        int c = col0 + wn * 64 + nt * 16 + lx;    // < 384
        float bp = bproj[c];
        #pragma unroll
        for (int mtt = 0; mtt < 2; ++mtt) {
            #pragma unroll
            for (int j = 0; j < 8; ++j) {
                int gr = row0 + wm * 32 + mtt * 16 + j + lh * 8;
                out[(size_t)gr * XDIM + c] = acc[mtt][nt][j] + bp;
            }
        }
    }
}

// ---------------------------------------------------------------------------
extern "C" void kernel_launch(void* const* d_in, const int* in_sizes, int n_in,
                              void* d_out, int out_size, void* d_ws, size_t ws_size,
                              hipStream_t stream) {
    const float* x     = (const float*)d_in[0];
    const float* Wqkv  = (const float*)d_in[1];
    const float* bqkv  = (const float*)d_in[2];
    const float* Wproj = (const float*)d_in[3];
    const float* bproj = (const float*)d_in[4];
    const float* ab    = (const float*)d_in[5];
    const int*   idxs  = (const int*)d_in[6];
    float* out = (float*)d_out;

    char* ws = (char*)d_ws;
    unsigned short* qb = (unsigned short*)(ws + Q_OFF);
    unsigned short* kb = (unsigned short*)(ws + K_OFF);
    unsigned short* vb = (unsigned short*)(ws + V_OFF);
    unsigned short* ao = (unsigned short*)(ws + AO_OFF);
    float*          bs = (float*)(ws + B_OFF);
    int n_off = in_sizes[5] / HEADS;

    // zero padded V region so padded P columns (p==0) can't meet NaN garbage
    (void)hipMemsetAsync(vb, 0, V_BYTES, stream);

    bias_kernel<<<(NTOK * NTOK + 255) / 256, 256, 0, stream>>>(ab, idxs, bs, n_off);
    qkv_kernel<<<dim3(ROWS / 128, HID / 128), 256, 0, stream>>>(x, Wqkv, bqkv, qb, kb, vb);
    attn_kernel<<<(2048 * 13) / 8, 256, 0, stream>>>(qb, kb, vb, bs, ao);
    proj_kernel<<<dim3(ROWS / 128, XDIM / 128), 256, 0, stream>>>(ao, Wproj, bproj, out);
}
